// PolicyNet_81930796139048
// MI455X (gfx1250) — compile-verified, matrix operand order checked by reading, not run on hardware
//
#include <hip/hip_runtime.h>

#define NNODES 50000
#define NEDGES 625000
#define DHID   128
#define DOUTSZ 32
#define NEG_SLOPE 0.2f

typedef __attribute__((ext_vector_type(16))) _Float16 v16h;
typedef __attribute__((ext_vector_type(8)))  _Float16 v8h;
typedef __attribute__((ext_vector_type(8)))  float    v8f;

static inline int cdiv(int a, int b) { return (a + b - 1) / b; }

// ---------------- elementwise utility kernels ----------------

__global__ void zero_f32(float* __restrict__ p, int n) {
  int i = blockIdx.x * blockDim.x + threadIdx.x;
  if (i < n) p[i] = 0.0f;
}

// Wt[n*K + k] = W[k*N + n]  (weights stored (K=din, N=dout) row-major in inputs)
__global__ void wtrans_f16(const float* __restrict__ W, _Float16* __restrict__ Wt,
                           int K, int N) {
  int i = blockIdx.x * blockDim.x + threadIdx.x;
  if (i >= K * N) return;
  int k = i / N, n = i % N;
  Wt[(size_t)n * K + k] = (_Float16)W[i];
}

// out[i] = act(out[i] + bias[i % N])
__global__ void bias_act(float* __restrict__ o, const float* __restrict__ b,
                         int n, int N, int relu) {
  int i = blockIdx.x * blockDim.x + threadIdx.x;
  if (i >= n) return;
  float v = o[i] + b[i % N];
  if (relu) v = v > 0.0f ? v : 0.0f;
  o[i] = v;
}

// ---------------- WMMA GEMM: C[M,N] = A[M,K](fp32, cvt in-reg) * Bt[N,K]^T(f16) + bias
// One wave32 per 16-row x N-column output slab. Per K-step: one A fragment
// (fp32 load + in-reg cvt) reused across all NT column tiles; all NT B
// fragments are batch-loaded into live registers FIRST (one s_clause, one
// wait), then NT back-to-back v_wmma. acc starts at 0; bias+relu fused into
// the store epilogue. ~160 VGPRs for NT=8.
template <int K, int NT>
__global__ void gemm_wmma_f16(const float* __restrict__ A,
                              const _Float16* __restrict__ Bt,
                              const float* __restrict__ bias,
                              float* __restrict__ C,
                              int M, int relu) {
  constexpr int N = NT * 16;
  const int lane = threadIdx.x & 31;
  const int wid = blockIdx.x * (blockDim.x >> 5) + (threadIdx.x >> 5);
  if (wid * 16 >= M) return;  // whole-wave uniform exit: EXEC all-ones for WMMA

  const int row0 = wid * 16;
  const int half = lane >> 4;  // which 16-lane half
  const int idx  = lane & 15;

  v8f acc[NT];
#pragma unroll
  for (int nt = 0; nt < NT; ++nt)
#pragma unroll
    for (int r = 0; r < 8; ++r) acc[nt][r] = 0.0f;

  // A fragment (16x32, M x K): lane<16 row M=idx holds K {0..7,16..23}+kk;
  // lane>=16 same rows, K {8..15,24..31}+kk.
  const float*    arow  = A  + (size_t)(row0 + idx) * K + half * 8;
  // B fragment (32x16, K x N) from N-major Bt: mirrored layout, col = nt*16+idx.
  const _Float16* bbase = Bt + (size_t)idx * K + half * 8;

#pragma unroll
  for (int kk = 0; kk < K; kk += 32) {
    // A fragment: 8 fp32 + 8 fp32, convert to f16 in registers
    const float4 a0 = *(const float4*)(arow + kk);
    const float4 a1 = *(const float4*)(arow + kk + 4);
    const float4 a2 = *(const float4*)(arow + kk + 16);
    const float4 a3 = *(const float4*)(arow + kk + 20);
    v16h av;
    av[0]  = (_Float16)a0.x; av[1]  = (_Float16)a0.y;
    av[2]  = (_Float16)a0.z; av[3]  = (_Float16)a0.w;
    av[4]  = (_Float16)a1.x; av[5]  = (_Float16)a1.y;
    av[6]  = (_Float16)a1.z; av[7]  = (_Float16)a1.w;
    av[8]  = (_Float16)a2.x; av[9]  = (_Float16)a2.y;
    av[10] = (_Float16)a2.z; av[11] = (_Float16)a2.w;
    av[12] = (_Float16)a3.x; av[13] = (_Float16)a3.y;
    av[14] = (_Float16)a3.z; av[15] = (_Float16)a3.w;

    // Batch-load ALL B fragments for this K-step (keeps 2*NT loads in flight)
    v16h bf[NT];
#pragma unroll
    for (int nt = 0; nt < NT; ++nt) {
      const _Float16* bp = bbase + (size_t)nt * 16 * K + kk;
      const v8h b0 = *(const v8h*)(bp);
      const v8h b1 = *(const v8h*)(bp + 16);
#pragma unroll
      for (int i = 0; i < 8; ++i) { bf[nt][i] = b0[i]; bf[nt][i + 8] = b1[i]; }
    }

    // Back-to-back WMMA chain over the column tiles
#pragma unroll
    for (int nt = 0; nt < NT; ++nt) {
      acc[nt] = __builtin_amdgcn_wmma_f32_16x16x32_f16(
          /*neg_a=*/false, av, /*neg_b=*/false, bf[nt],
          /*c_mod=*/(short)0, acc[nt], /*reuse_a=*/false, /*reuse_b=*/false);
    }
  }

#pragma unroll
  for (int nt = 0; nt < NT; ++nt) {
    const float bv = bias[nt * 16 + idx];
    float* crow = C + (size_t)(row0 + half * 8) * N + nt * 16 + idx;
#pragma unroll
    for (int r = 0; r < 8; ++r) {
      float v = acc[nt][r] + bv;
      if (relu) v = v > 0.0f ? v : 0.0f;
      crow[(size_t)r * N] = v;
    }
  }
}

// ---------------- edge phase (L2-resident gathers; wave32 per edge) ----------------

// e[i] = dot(leakyrelu(xl[src] + xr[dst]), att);  m_ord[dst] = atomic max (ordered-uint)
__global__ void edge_logits_kernel(const float* __restrict__ xl,
                                   const float* __restrict__ xr,
                                   const long long* __restrict__ src,
                                   const long long* __restrict__ dst,
                                   const float* __restrict__ att,
                                   float* __restrict__ e,
                                   unsigned int* __restrict__ m_ord,
                                   int E) {
  const int lane = threadIdx.x & 31;
  const int edge = blockIdx.x * (blockDim.x >> 5) + (threadIdx.x >> 5);
  if (edge >= E) return;
  const long long s = src[edge];
  const long long d = dst[edge];
  const float4 a = *(const float4*)(xl + (size_t)s * DHID + lane * 4);
  const float4 b = *(const float4*)(xr + (size_t)d * DHID + lane * 4);
  const float4 w = *(const float4*)(att + lane * 4);
  float p = 0.0f, h;
  h = a.x + b.x; h = h > 0.0f ? h : NEG_SLOPE * h; p += h * w.x;
  h = a.y + b.y; h = h > 0.0f ? h : NEG_SLOPE * h; p += h * w.y;
  h = a.z + b.z; h = h > 0.0f ? h : NEG_SLOPE * h; p += h * w.z;
  h = a.w + b.w; h = h > 0.0f ? h : NEG_SLOPE * h; p += h * w.w;
#pragma unroll
  for (int m = 16; m >= 1; m >>= 1) p += __shfl_xor(p, m, 32);
  if (lane == 0) {
    e[edge] = p;
    unsigned bits = __float_as_uint(p);
    unsigned ord = (bits & 0x80000000u) ? ~bits : (bits | 0x80000000u);
    atomicMax(&m_ord[d], ord);
  }
}

// ee = exp(e - m[dst]); denom[dst] += ee  (e buffer overwritten with ee)
__global__ void edge_exp_kernel(float* __restrict__ e,
                                const long long* __restrict__ dst,
                                const unsigned int* __restrict__ m_ord,
                                float* __restrict__ denom, int E) {
  int i = blockIdx.x * blockDim.x + threadIdx.x;
  if (i >= E) return;
  long long d = dst[i];
  unsigned ord = m_ord[d];
  unsigned bits = (ord & 0x80000000u) ? (ord & 0x7FFFFFFFu) : ~ord;
  float m = __uint_as_float(bits);
  if (!__builtin_isfinite(m)) m = 0.0f;
  float ee = __expf(e[i] - m);
  e[i] = ee;
  atomicAdd(&denom[d], ee);
}

// out[dst] += (ee / (denom[dst]+1e-16)) * xl[src]
__global__ void edge_scatter_kernel(const float* __restrict__ xl,
                                    const long long* __restrict__ src,
                                    const long long* __restrict__ dst,
                                    const float* __restrict__ ee,
                                    const float* __restrict__ denom,
                                    float* __restrict__ out, int E) {
  const int lane = threadIdx.x & 31;
  const int edge = blockIdx.x * (blockDim.x >> 5) + (threadIdx.x >> 5);
  if (edge >= E) return;
  const long long s = src[edge];
  const long long d = dst[edge];
  const float alpha = ee[edge] / (denom[d] + 1e-16f);
  const float4 a = *(const float4*)(xl + (size_t)s * DHID + lane * 4);
  float* o = out + (size_t)d * DHID + lane * 4;
  atomicAdd(o + 0, alpha * a.x);
  atomicAdd(o + 1, alpha * a.y);
  atomicAdd(o + 2, alpha * a.z);
  atomicAdd(o + 3, alpha * a.w);
}

// ---------------- host orchestration ----------------

extern "C" void kernel_launch(void* const* d_in, const int* in_sizes, int n_in,
                              void* d_out, int out_size, void* d_ws, size_t ws_size,
                              hipStream_t stream) {
  const float* x       = (const float*)d_in[0];
  const long long* ei  = (const long long*)d_in[1];
  const long long* src = ei;
  const long long* dst = ei + NEDGES;

  const float* Wl[3]  = {(const float*)d_in[2],  (const float*)d_in[8],  (const float*)d_in[14]};
  const float* bl[3]  = {(const float*)d_in[3],  (const float*)d_in[9],  (const float*)d_in[15]};
  const float* Wr[3]  = {(const float*)d_in[4],  (const float*)d_in[10], (const float*)d_in[16]};
  const float* br[3]  = {(const float*)d_in[5],  (const float*)d_in[11], (const float*)d_in[17]};
  const float* att[3] = {(const float*)d_in[6],  (const float*)d_in[12], (const float*)d_in[18]};
  const float* bs[3]  = {(const float*)d_in[7],  (const float*)d_in[13], (const float*)d_in[19]};
  const float* Wfc1 = (const float*)d_in[20]; const float* bfc1 = (const float*)d_in[21];
  const float* Wfc2 = (const float*)d_in[22]; const float* bfc2 = (const float*)d_in[23];
  const float* Wfc3 = (const float*)d_in[24]; const float* bfc3 = (const float*)d_in[25];

  // workspace carve-up (256B aligned)
  char* w = (char*)d_ws;
  size_t off = 0;
  auto carve = [&](size_t bytes) -> void* {
    off = (off + 255) & ~(size_t)255;
    void* p = w + off;
    off += bytes;
    return p;
  };
  float*     xl    = (float*)carve((size_t)NNODES * DHID * 4);
  float*     xr    = (float*)carve((size_t)NNODES * DHID * 4);
  float*     outb  = (float*)carve((size_t)NNODES * DHID * 4);
  _Float16*  Wt    = (_Float16*)carve((size_t)DHID * DHID * 2);
  float*     ebuf  = (float*)carve((size_t)NEDGES * 4);
  unsigned*  mord  = (unsigned*)carve((size_t)NNODES * 4);
  float*     denom = (float*)carve((size_t)NNODES * 4);

  const int nelem = NNODES * DHID;
  const int gemm_waves = NNODES / 16;              // 3125 row-slabs, wave each
  const int gemm_grid  = cdiv(gemm_waves, 8);      // 8 waves / 256-thread block
  const int edge_wgrid = cdiv(NEDGES, 8);          // wave-per-edge kernels

  // ---- 3 GATv2 layers ----
  const float* in = x;
  for (int li = 0; li < 3; ++li) {
    wtrans_f16<<<cdiv(DHID * DHID, 256), 256, 0, stream>>>(Wl[li], Wt, DHID, DHID);
    gemm_wmma_f16<DHID, DHID / 16><<<gemm_grid, 256, 0, stream>>>(in, Wt, bl[li], xl,
                                                                  NNODES, 0);
    wtrans_f16<<<cdiv(DHID * DHID, 256), 256, 0, stream>>>(Wr[li], Wt, DHID, DHID);
    gemm_wmma_f16<DHID, DHID / 16><<<gemm_grid, 256, 0, stream>>>(in, Wt, br[li], xr,
                                                                  NNODES, 0);

    zero_f32<<<cdiv(nelem, 256), 256, 0, stream>>>(outb, nelem);
    zero_f32<<<cdiv(NNODES, 256), 256, 0, stream>>>((float*)mord, NNODES);
    zero_f32<<<cdiv(NNODES, 256), 256, 0, stream>>>(denom, NNODES);

    edge_logits_kernel<<<edge_wgrid, 256, 0, stream>>>(xl, xr, src, dst, att[li],
                                                       ebuf, mord, NEDGES);
    edge_exp_kernel<<<cdiv(NEDGES, 256), 256, 0, stream>>>(ebuf, dst, mord, denom,
                                                           NEDGES);
    edge_scatter_kernel<<<edge_wgrid, 256, 0, stream>>>(xl, src, dst, ebuf, denom,
                                                        outb, NEDGES);
    bias_act<<<cdiv(nelem, 256), 256, 0, stream>>>(outb, bs[li], nelem, DHID,
                                                   (li < 2) ? 1 : 0);
    in = outb;
  }

  // ---- MLP head ----
  wtrans_f16<<<cdiv(DHID * DHID, 256), 256, 0, stream>>>(Wfc1, Wt, DHID, DHID);
  gemm_wmma_f16<DHID, DHID / 16><<<gemm_grid, 256, 0, stream>>>(outb, Wt, bfc1, xl,
                                                                NNODES, 1);

  wtrans_f16<<<cdiv(DHID * DHID, 256), 256, 0, stream>>>(Wfc2, Wt, DHID, DHID);
  gemm_wmma_f16<DHID, DHID / 16><<<gemm_grid, 256, 0, stream>>>(xl, Wt, bfc2, xr,
                                                                NNODES, 1);

  wtrans_f16<<<cdiv(DHID * DOUTSZ, 256), 256, 0, stream>>>(Wfc3, Wt, DHID, DOUTSZ);
  gemm_wmma_f16<DHID, DOUTSZ / 16><<<gemm_grid, 256, 0, stream>>>(
      xr, Wt, bfc3, (float*)d_out, NNODES, 0);
}
